// BilinearModelTopK_49830210568807
// MI455X (gfx1250) — compile-verified, matrix operand order checked by reading, not run on hardware
//
#include <hip/hip_runtime.h>
#include <hip/hip_bf16.h>

typedef __attribute__((ext_vector_type(16))) __bf16 v16bf;
typedef __attribute__((ext_vector_type(8)))  float  v8f;
typedef __attribute__((ext_vector_type(4)))  float  vf4;

#define BATCH  64
#define DFULL  1024
#define DD     512
#define OUTD   1024

// Map (e,b) -> position in the 16-bit 32x16 WMMA B-operand fragment stream.
// Fragment (ek = e/32, bt = b/16): lane = (b&15) + 16*((e&31)>>3 & 1),
// slot = (e&7) + 8*((e&31)>>4). Each lane's 16 bf16 are stored contiguously
// (32B) so a fragment load is one 32-byte LDS read per lane.
__device__ __forceinline__ int fragpos(int e, int b) {
  int ek   = e >> 5;
  int el   = e & 31;
  int half = (el >> 3) & 1;
  int slot = (el & 7) + ((el >> 4) << 3);
  int bt   = b >> 4;
  int lane = (b & 15) + (half << 4);
  return ((ek * 4 + bt) * 32 + lane) * 16 + slot;
}

__device__ __forceinline__ v16bf cvt16(vf4 q0, vf4 q1, vf4 q2, vf4 q3) {
  v16bf v;
  v[0]  = (__bf16)q0[0]; v[1]  = (__bf16)q0[1]; v[2]  = (__bf16)q0[2]; v[3]  = (__bf16)q0[3];
  v[4]  = (__bf16)q1[0]; v[5]  = (__bf16)q1[1]; v[6]  = (__bf16)q1[2]; v[7]  = (__bf16)q1[3];
  v[8]  = (__bf16)q2[0]; v[9]  = (__bf16)q2[1]; v[10] = (__bf16)q2[2]; v[11] = (__bf16)q2[3];
  v[12] = (__bf16)q3[0]; v[13] = (__bf16)q3[1]; v[14] = (__bf16)q3[2]; v[15] = (__bf16)q3[3];
  return v;
}

// K0: unit-L2 normalize x rows, gather columns by input_idxs, emit
//  - xg_t[e][b] fp32 (for the fused quadratic-form contraction)
//  - xfrag      bf16 (pre-swizzled WMMA B-operand fragments, 64KB)
__global__ void k_norm_gather(const float* __restrict__ x, const int* __restrict__ idxs,
                              float* __restrict__ xg_t, __bf16* __restrict__ xfrag) {
  __shared__ float red[256];
  int b = blockIdx.x, tid = threadIdx.x;
  const float* xr = x + (size_t)b * DFULL;
  float ssq = 0.f;
  for (int j = tid; j < DFULL; j += 256) { float v = xr[j]; ssq += v * v; }
  red[tid] = ssq; __syncthreads();
  for (int s = 128; s > 0; s >>= 1) { if (tid < s) red[tid] += red[tid + s]; __syncthreads(); }
  float rn = rsqrtf(red[0]);
  for (int e = tid; e < DD; e += 256) {
    float v = xr[idxs[e]] * rn;
    xg_t[(size_t)e * BATCH + b] = v;
    xfrag[fragpos(e, b)] = (__bf16)v;
  }
}

// K2: normalize y1 columns-per-batch (y stored [s][b]), emit xg_t + xfrag for layer 2.
__global__ void k_norm_cols(const float* __restrict__ yt,
                            float* __restrict__ xg_t, __bf16* __restrict__ xfrag) {
  __shared__ float red[128];
  int b = blockIdx.x, tid = threadIdx.x;
  float ssq = 0.f;
  for (int s = tid; s < DD; s += 128) { float v = yt[(size_t)s * BATCH + b]; ssq += v * v; }
  red[tid] = ssq; __syncthreads();
  for (int s = 64; s > 0; s >>= 1) { if (tid < s) red[tid] += red[tid + s]; __syncthreads(); }
  float rn = rsqrtf(red[0]);
  for (int e = tid; e < DD; e += 128) {
    float v = yt[(size_t)e * BATCH + b] * rn;
    xg_t[(size_t)e * BATCH + b] = v;
    xfrag[fragpos(e, b)] = (__bf16)v;
  }
}

// K1/K3: fused bilinear y[b,s] = x_b^T B[s] x_b for one s per workgroup.
// 8 waves; wave owns 4 d-tiles. H[d,b] tiles built by v_wmma_f32_16x16x32_bf16
// over 16 e-chunks; B[s] streamed once from HBM with NT loads (fp32->bf16 in
// registers); x fragments served from LDS; H contracted against x[d,b] on the
// fly, reduced via LDS float atomics.
__global__ void __launch_bounds__(256)
k_bilinear(const float* __restrict__ Bt,              // (512,512,512) fp32
           const unsigned char* __restrict__ xfrag_g, // 64KB bf16 fragments
           const float* __restrict__ xg_t,            // [d][b] fp32
           float* __restrict__ yt) {                  // [s][b] fp32 out
  extern __shared__ unsigned char smem[];             // 64KB frags + 256B reduce
  float* ylds = (float*)(smem + 65536);
  int tid = threadIdx.x;
  int s = blockIdx.x;

  { // cooperative copy of pre-swizzled x fragments into LDS (16B vectors)
    const vf4* src = (const vf4*)xfrag_g;
    vf4* dst = (vf4*)smem;
    for (int i = tid; i < 4096; i += 256) dst[i] = src[i];
  }
  if (tid < BATCH) ylds[tid] = 0.f;
  __syncthreads();

  int wave = tid >> 5;
  int lane = tid & 31;
  int bl = lane & 15;   // row-in-tile / batch-col-in-tile
  int h  = lane >> 4;   // lane half (K interleave per ISA 16-bit A/B layout)

  const float* Bs = Bt + (size_t)s * (DD * DD);
  float ysum[4] = {0.f, 0.f, 0.f, 0.f};

  for (int dti = 0; dti < 4; ++dti) {
    int dt = wave * 4 + dti;                      // d-tile 0..31
    const float* rowp = Bs + (size_t)(dt * 16 + bl) * DD + h * 8;
    v8f acc0 = {}, acc1 = {}, acc2 = {}, acc3 = {};
    for (int ek = 0; ek < 16; ++ek) {
      const float* rp = rowp + ek * 32;
      // A-operand: lane (M = bl) needs K = {h*8+0..7, 16+h*8+0..7} of this e-chunk
      vf4 q0 = __builtin_nontemporal_load((const vf4*)rp);
      vf4 q1 = __builtin_nontemporal_load((const vf4*)(rp + 4));
      vf4 q2 = __builtin_nontemporal_load((const vf4*)(rp + 16));
      vf4 q3 = __builtin_nontemporal_load((const vf4*)(rp + 20));
      v16bf a = cvt16(q0, q1, q2, q3);
      const v16bf* xf = (const v16bf*)(smem + (size_t)(ek * 4) * 32 * 32);
      v16bf b0 = xf[lane];
      v16bf b1 = xf[32 + lane];
      v16bf b2 = xf[64 + lane];
      v16bf b3 = xf[96 + lane];
      acc0 = __builtin_amdgcn_wmma_f32_16x16x32_bf16(false, a, false, b0, (short)0, acc0, false, false);
      acc1 = __builtin_amdgcn_wmma_f32_16x16x32_bf16(false, a, false, b1, (short)0, acc1, false, false);
      acc2 = __builtin_amdgcn_wmma_f32_16x16x32_bf16(false, a, false, b2, (short)0, acc2, false, false);
      acc3 = __builtin_amdgcn_wmma_f32_16x16x32_bf16(false, a, false, b3, (short)0, acc3, false, false);
    }
    // C/D layout: vgpr r, lane -> (M = r + 8*h, N = bl). Contract H*x over d.
#pragma unroll
    for (int r = 0; r < 8; ++r) {
      int d = dt * 16 + r + 8 * h;
      const float* xp = xg_t + (size_t)d * BATCH;
      ysum[0] += acc0[r] * xp[bl];
      ysum[1] += acc1[r] * xp[16 + bl];
      ysum[2] += acc2[r] * xp[32 + bl];
      ysum[3] += acc3[r] * xp[48 + bl];
    }
  }
  atomicAdd(&ylds[bl],      ysum[0]);
  atomicAdd(&ylds[16 + bl], ysum[1]);
  atomicAdd(&ylds[32 + bl], ysum[2]);
  atomicAdd(&ylds[48 + bl], ysum[3]);
  __syncthreads();
  if (tid < BATCH) yt[(size_t)s * BATCH + tid] = ylds[tid];
}

// K4: out[b,o] = sum_s y2[b,s] * W[o,s] + bias[o].  M=64, N=1024, K=512.
// 256 waves, one 16x16 output tile each, 16 K-chunks of 32 via bf16 WMMA.
__global__ void __launch_bounds__(256)
k_out_gemm(const float* __restrict__ yt,   // y2 as [s][b]
           const float* __restrict__ W,    // [o][s]
           const float* __restrict__ bias, // [o]
           float* __restrict__ out) {      // [b][o]
  int tid = threadIdx.x;
  int wave = tid >> 5, lane = tid & 31;
  int bl = lane & 15, h = lane >> 4;
  int tile = blockIdx.x * 8 + wave;        // 0..255
  int mt = tile & 3;                       // batch tile (4)
  int nt = tile >> 2;                      // output tile (64)
  int b_row = mt * 16 + bl;                // A operand M
  int o_col = nt * 16 + bl;                // B operand N
  const float* wp = W + (size_t)o_col * DD + h * 8;
  v8f acc = {};
  for (int kc = 0; kc < 16; ++kc) {
    int k0 = kc * 32 + h * 8;
    v16bf a;
#pragma unroll
    for (int j = 0; j < 8; ++j) {
      a[j]     = (__bf16)yt[(size_t)(k0 + j) * BATCH + b_row];
      a[j + 8] = (__bf16)yt[(size_t)(k0 + 16 + j) * BATCH + b_row];
    }
    const float* rp = wp + kc * 32;
    vf4 q0 = *(const vf4*)rp;
    vf4 q1 = *(const vf4*)(rp + 4);
    vf4 q2 = *(const vf4*)(rp + 16);
    vf4 q3 = *(const vf4*)(rp + 20);
    v16bf w = cvt16(q0, q1, q2, q3);
    acc = __builtin_amdgcn_wmma_f32_16x16x32_bf16(false, a, false, w, (short)0, acc, false, false);
  }
  float bo = bias[o_col];
#pragma unroll
  for (int r = 0; r < 8; ++r) {
    int bb = mt * 16 + r + 8 * h;
    out[(size_t)bb * OUTD + o_col] = acc[r] + bo;
  }
}

extern "C" void kernel_launch(void* const* d_in, const int* in_sizes, int n_in,
                              void* d_out, int out_size, void* d_ws, size_t ws_size,
                              hipStream_t stream) {
  (void)in_sizes; (void)n_in; (void)out_size; (void)ws_size;
  const float* x    = (const float*)d_in[0];
  const float* B1   = (const float*)d_in[1];
  const float* B2   = (const float*)d_in[2];
  const float* Wout = (const float*)d_in[3];
  const float* bias = (const float*)d_in[4];
  const int*   idxs = (const int*)d_in[5];
  float* out = (float*)d_out;

  char* ws = (char*)d_ws;                        // 640KB used
  __bf16* xfrag1 = (__bf16*)(ws + (0   << 10));  // 64KB  swizzled bf16 frags
  float*  xg1    = (float*) (ws + (64  << 10));  // 128KB xg_t [d][b] fp32
  float*  y1t    = (float*) (ws + (192 << 10));  // 128KB y1 [s][b]
  __bf16* xfrag2 = (__bf16*)(ws + (320 << 10));  // 64KB
  float*  xg2    = (float*) (ws + (384 << 10));  // 128KB
  float*  y2t    = (float*) (ws + (512 << 10));  // 128KB

  const size_t dynlds = 65536 + 64 * sizeof(float);

  k_norm_gather<<<64, 256, 0, stream>>>(x, idxs, xg1, xfrag1);
  k_bilinear<<<512, 256, dynlds, stream>>>(B1, (const unsigned char*)xfrag1, xg1, y1t);
  k_norm_cols<<<64, 128, 0, stream>>>(y1t, xg2, xfrag2);
  k_bilinear<<<512, 256, dynlds, stream>>>(B2, (const unsigned char*)xfrag2, xg2, y2t);
  k_out_gemm<<<32, 256, 0, stream>>>(y2t, Wout, bias, out);
}
